// Seq_HyGAN_89111981457968
// MI455X (gfx1250) — matrix operand order repaired and checked by reading, not compiled, not run on hardware
//
#include <hip/hip_runtime.h>
#include <hip/hip_bf16.h>

typedef __attribute__((ext_vector_type(16))) _Float16 v16h;
typedef __attribute__((ext_vector_type(8)))  float    v8f;

#define D_DIM 128
#define GEMM_WAVES 8   // 8 waves (256 threads) per block, one 16-row tile per wave

// ---------------- monotone float <-> u32 order-preserving encode -------------
__device__ __forceinline__ unsigned encf(float f) {
    unsigned u = __float_as_uint(f);
    return (u & 0x80000000u) ? ~u : (u | 0x80000000u);
}
__device__ __forceinline__ float decf(unsigned u) {
    return __uint_as_float((u & 0x80000000u) ? (u ^ 0x80000000u) : ~u);
}

// ---------------- fills ------------------------------------------------------
__global__ void fill_f32(float* __restrict__ p, float v, long long n) {
    long long i = (long long)blockIdx.x * blockDim.x + threadIdx.x;
    long long s = (long long)gridDim.x * blockDim.x;
    for (; i < n; i += s) p[i] = v;
}
__global__ void fill_u32(unsigned* __restrict__ p, unsigned v, long long n) {
    long long i = (long long)blockIdx.x * blockDim.x + threadIdx.x;
    long long s = (long long)gridDim.x * blockDim.x;
    for (; i < n; i += s) p[i] = v;
}

// ------------- weight convert: W[k][n] f32 -> Wt[n][k] f16 -------------------
__global__ void convert_w(const float* __restrict__ W, _Float16* __restrict__ Wt) {
    int i = blockIdx.x * blockDim.x + threadIdx.x;
    if (i >= D_DIM * D_DIM) return;
    int n = i >> 7, k = i & 127;
    Wt[i] = (_Float16)W[k * D_DIM + n];
}

// ------------- WMMA GEMM: Y[rows,128] = X[rows,128] @ W + b ------------------
// Wt is f16, transposed [n][k]. Weights staged once per block into LDS;
// each wave owns one 16-row tile and all 8 column tiles.
__global__ __launch_bounds__(GEMM_WAVES * 32) void gemm128_wmma(
    const float* __restrict__ X, const _Float16* __restrict__ Wt,
    const float* __restrict__ bias, float* __restrict__ Y, int rowTiles)
{
    __shared__ _Float16 sW[D_DIM * D_DIM];   // 32 KB of 320 KB WGP LDS

    // ---- stage full weight matrix to LDS (b128 copies) ----
    {
        const uint4* src = (const uint4*)Wt;
        uint4* dst = (uint4*)sW;
        const int nVec = D_DIM * D_DIM * (int)sizeof(_Float16) / 16;  // 2048
        #pragma unroll
        for (int i = threadIdx.x; i < nVec; i += GEMM_WAVES * 32) dst[i] = src[i];
    }
    __syncthreads();

    const int lane   = threadIdx.x & 31;
    const int wave   = threadIdx.x >> 5;
    const int tile   = blockIdx.x * GEMM_WAVES + wave;
    if (tile >= rowTiles) return;          // wave-uniform: EXEC stays all-1s
    const int lane15 = lane & 15;
    const int khalf  = lane >> 4;
    const float* xrow = X + (size_t)(tile * 16 + lane15) * D_DIM;

    v8f acc[8] = {};

    #pragma unroll
    for (int kb = 0; kb < D_DIM; kb += 32) {
        // ---- A fragment (16x32 f16, ISA 16-bit A layout) from global ----
        v16h a;
        #pragma unroll
        for (int j = 0; j < 8; ++j) {
            int k = kb + ((j & 4) ? 16 : 0) + 8 * khalf + 2 * (j & 3);
            float2 f = *(const float2*)(xrow + k);
            a[2 * j]     = (_Float16)f.x;
            a[2 * j + 1] = (_Float16)f.y;
        }
        // ---- 8 column tiles, B fragments from LDS (ds_load_b128) ----
        #pragma unroll
        for (int t = 0; t < 8; ++t) {
            const _Float16* wcol = sW + (t * 16 + lane15) * D_DIM;
            v16h b;
            #pragma unroll
            for (int j = 0; j < 8; ++j) {
                int k = kb + ((j & 4) ? 16 : 0) + 8 * khalf + 2 * (j & 3);   // two 8-half runs/lane (table order)
                b[2 * j]     = wcol[k];
                b[2 * j + 1] = wcol[k + 1];
            }
            acc[t] = __builtin_amdgcn_wmma_f32_16x16x32_f16(
                false, a, false, b, (short)0, acc[t], false, false);
        }
    }

    // ---- store D (f32 16x16 C/D layout) + bias ----
    #pragma unroll
    for (int t = 0; t < 8; ++t) {
        int col = t * 16 + lane15;
        float bv = bias[col];
        #pragma unroll
        for (int r = 0; r < 8; ++r) {
            int row = tile * 16 + r + 8 * khalf;
            Y[(size_t)row * D_DIM + col] = acc[t][r] + bv;
        }
    }
}

// ------------- edge attention logits + segment max ---------------------------
// attn[e] = leaky_relu(dot(A[idxA[e]], B[idxB[e]])) * scale ; atomicMax per seg
__global__ __launch_bounds__(256) void edge_attn(
    const float* __restrict__ A, const int* __restrict__ idxA,
    const float* __restrict__ B, const int* __restrict__ idxB,
    const int* __restrict__ seg, float* __restrict__ attn,
    unsigned* __restrict__ segmax, int E, float scale)
{
    int lane = threadIdx.x & 31;
    int e = blockIdx.x * 8 + (threadIdx.x >> 5);
    if (e >= E) return;
    const float4* a4 = (const float4*)(A + (size_t)idxA[e] * D_DIM);
    const float4* b4 = (const float4*)(B + (size_t)idxB[e] * D_DIM);
    float4 x = a4[lane], y = b4[lane];
    float s = x.x * y.x + x.y * y.y + x.z * y.z + x.w * y.w;
    #pragma unroll
    for (int off = 16; off; off >>= 1) s += __shfl_xor(s, off, 32);
    if (lane == 0) {
        s = (s > 0.0f ? s : 0.01f * s) * scale;   // leaky_relu then scale
        attn[e] = s;
        atomicMax(&segmax[seg[e]], encf(s));
    }
}

// ------------- exp(logit - segmax) and segment sum ---------------------------
__global__ void edge_expsum(float* __restrict__ attn, const int* __restrict__ seg,
                            const unsigned* __restrict__ segmax,
                            float* __restrict__ segsum, int E)
{
    int e = blockIdx.x * blockDim.x + threadIdx.x;
    if (e >= E) return;
    int sg = seg[e];
    float v = __expf(attn[e] - decf(segmax[sg]));
    attn[e] = v;
    atomicAdd(&segsum[sg], v);
}

// ------------- weighted scatter: out[seg] += (e/sum) * V[src] ----------------
__global__ __launch_bounds__(256) void edge_scatter(
    const float* __restrict__ attn, const float* __restrict__ segsum,
    const float* __restrict__ V, const int* __restrict__ src,
    const int* __restrict__ seg, float* __restrict__ out, int E)
{
    int lane = threadIdx.x & 31;
    int e = blockIdx.x * 8 + (threadIdx.x >> 5);
    if (e >= E) return;
    int sg = seg[e];
    float ssum = segsum[sg];
    float w = attn[e] / (ssum > 0.0f ? ssum : 1.0f);
    const float4* v4 = (const float4*)(V + (size_t)src[e] * D_DIM);
    float4 v = v4[lane];
    float* o = out + (size_t)sg * D_DIM + lane * 4;
    atomicAdd(o + 0, w * v.x);
    atomicAdd(o + 1, w * v.y);
    atomicAdd(o + 2, w * v.z);
    atomicAdd(o + 3, w * v.w);
}

// ------------- classification head: pred[M,10] = F @ Wc + bc -----------------
__global__ void head10(const float* __restrict__ F, const float* __restrict__ Wc,
                       const float* __restrict__ bc, float* __restrict__ out, int M)
{
    int i = blockIdx.x * blockDim.x + threadIdx.x;
    if (i >= M * 10) return;
    int m = i / 10, c = i % 10;
    const float* f = F + (size_t)m * D_DIM;
    float s = bc[c];
    #pragma unroll 8
    for (int k = 0; k < D_DIM; ++k) s = fmaf(f[k], Wc[k * 10 + c], s);
    out[i] = s;
}

// ============================================================================
extern "C" void kernel_launch(void* const* d_in, const int* in_sizes, int n_in,
                              void* d_out, int out_size, void* d_ws, size_t ws_size,
                              hipStream_t stream)
{
    const float* vfeat  = (const float*)d_in[0];
    const float* efeat  = (const float*)d_in[1];
    const int*   he_idx = (const int*)d_in[2];
    const int*   nd_idx = (const int*)d_in[3];
    const float* Wfl = (const float*)d_in[6];  const float* bfl = (const float*)d_in[7];
    const float* W1  = (const float*)d_in[8];  const float* b1  = (const float*)d_in[9];
    const float* W2  = (const float*)d_in[10]; const float* b2  = (const float*)d_in[11];
    const float* W3  = (const float*)d_in[12]; const float* b3  = (const float*)d_in[13];
    const float* W4  = (const float*)d_in[14]; const float* b4  = (const float*)d_in[15];
    const float* W5  = (const float*)d_in[16]; const float* b5  = (const float*)d_in[17];
    const float* W6  = (const float*)d_in[18]; const float* b6  = (const float*)d_in[19];
    const float* Wc  = (const float*)d_in[20]; const float* bc  = (const float*)d_in[21];

    const int N = in_sizes[0] / D_DIM;   // 100000 nodes
    const int M = in_sizes[1] / D_DIM;   // 10000 hyperedges
    const int E = in_sizes[2];           // 640000 incidences
    const float scale = 0.08838834764831845f;  // 1/sqrt(128)

    // ---------------- workspace carve-up ----------------
    char* base = (char*)d_ws;
    size_t off = 0;
    auto alloc = [&](size_t bytes) -> void* {
        void* p = base + off;
        off += (bytes + 255) & ~(size_t)255;
        return p;
    };
    const size_t wBytes = (size_t)D_DIM * D_DIM * sizeof(_Float16);
    _Float16* hWfl = (_Float16*)alloc(wBytes);
    _Float16* hW1  = (_Float16*)alloc(wBytes);
    _Float16* hW2  = (_Float16*)alloc(wBytes);
    _Float16* hW3  = (_Float16*)alloc(wBytes);
    _Float16* hW4  = (_Float16*)alloc(wBytes);
    _Float16* hW5  = (_Float16*)alloc(wBytes);
    _Float16* hW6  = (_Float16*)alloc(wBytes);
    const size_t mRow = (size_t)M * D_DIM * sizeof(float);
    const size_t nRow = (size_t)N * D_DIM * sizeof(float);
    float* feat_e  = (float*)alloc(mRow);
    float* k_e     = (float*)alloc(mRow);
    float* v_e     = (float*)alloc(mRow);
    float* q_e     = (float*)alloc(mRow);
    float* feat_e2 = (float*)alloc(mRow);
    float* q_v     = (float*)alloc(nRow);   // reused as k_v after phase 1
    float* feat_v  = (float*)alloc(nRow);
    float* v_v     = (float*)alloc(nRow);
    float* attn    = (float*)alloc((size_t)E * sizeof(float));
    unsigned* segmax_v = (unsigned*)alloc((size_t)N * sizeof(unsigned));
    float*    segsum_v = (float*)alloc((size_t)N * sizeof(float));
    unsigned* segmax_e = (unsigned*)alloc((size_t)M * sizeof(unsigned));
    float*    segsum_e = (float*)alloc((size_t)M * sizeof(float));
    (void)ws_size; (void)n_in;

    // ---------------- weight conversion (f32 -> f16, transposed) ------------
    const int cwThreads = 256, cwBlocks = (D_DIM * D_DIM + cwThreads - 1) / cwThreads;
    convert_w<<<cwBlocks, cwThreads, 0, stream>>>(Wfl, hWfl);
    convert_w<<<cwBlocks, cwThreads, 0, stream>>>(W1,  hW1);
    convert_w<<<cwBlocks, cwThreads, 0, stream>>>(W2,  hW2);
    convert_w<<<cwBlocks, cwThreads, 0, stream>>>(W3,  hW3);
    convert_w<<<cwBlocks, cwThreads, 0, stream>>>(W4,  hW4);
    convert_w<<<cwBlocks, cwThreads, 0, stream>>>(W5,  hW5);
    convert_w<<<cwBlocks, cwThreads, 0, stream>>>(W6,  hW6);

    const int mTiles = M / 16, nTiles = N / 16;   // both divisible by 16
    const int mGemm = (mTiles + GEMM_WAVES - 1) / GEMM_WAVES;
    const int nGemm = (nTiles + GEMM_WAVES - 1) / GEMM_WAVES;
    const int gemmThr = GEMM_WAVES * 32;

    // ---------------- phase 0: projections ----------------
    gemm128_wmma<<<mGemm, gemmThr, 0, stream>>>(efeat,  hWfl, bfl, feat_e, mTiles);
    gemm128_wmma<<<mGemm, gemmThr, 0, stream>>>(feat_e, hW5,  b5,  k_e,    mTiles);
    gemm128_wmma<<<mGemm, gemmThr, 0, stream>>>(feat_e, hW6,  b6,  v_e,    mTiles);
    gemm128_wmma<<<nGemm, gemmThr, 0, stream>>>(vfeat,  hW4,  b4,  q_v,    nTiles);

    // ---------------- phase 1: hyperedge -> node attention -------------------
    fill_u32<<<2048, 256, 0, stream>>>(segmax_v, 0u,   (long long)N);
    fill_f32<<<2048, 256, 0, stream>>>(segsum_v, 0.0f, (long long)N);
    fill_f32<<<4096, 256, 0, stream>>>(feat_v,   0.0f, (long long)N * D_DIM);

    const int eWaveBlocks = (E + 7) / 8;
    const int eThrBlocks  = (E + 255) / 256;
    edge_attn<<<eWaveBlocks, 256, 0, stream>>>(k_e, he_idx, q_v, nd_idx, nd_idx,
                                               attn, segmax_v, E, scale);
    edge_expsum<<<eThrBlocks, 256, 0, stream>>>(attn, nd_idx, segmax_v, segsum_v, E);
    edge_scatter<<<eWaveBlocks, 256, 0, stream>>>(attn, segsum_v, v_e, he_idx,
                                                  nd_idx, feat_v, E);

    // ---------------- phase 2 projections ----------------
    float* k_v = q_v;  // q_v is dead after edge_attn of phase 1
    gemm128_wmma<<<nGemm, gemmThr, 0, stream>>>(feat_v, hW2, b2, k_v, nTiles);
    gemm128_wmma<<<nGemm, gemmThr, 0, stream>>>(feat_v, hW3, b3, v_v, nTiles);
    gemm128_wmma<<<mGemm, gemmThr, 0, stream>>>(feat_e, hW1, b1, q_e, mTiles);

    // ---------------- phase 2: node -> hyperedge attention -------------------
    fill_u32<<<512,  256, 0, stream>>>(segmax_e, 0u,   (long long)M);
    fill_f32<<<512,  256, 0, stream>>>(segsum_e, 0.0f, (long long)M);
    fill_f32<<<2048, 256, 0, stream>>>(feat_e2,  0.0f, (long long)M * D_DIM);

    edge_attn<<<eWaveBlocks, 256, 0, stream>>>(k_v, nd_idx, q_e, he_idx, he_idx,
                                               attn, segmax_e, E, scale);
    edge_expsum<<<eThrBlocks, 256, 0, stream>>>(attn, he_idx, segmax_e, segsum_e, E);
    edge_scatter<<<eWaveBlocks, 256, 0, stream>>>(attn, segsum_e, v_v, nd_idx,
                                                  he_idx, feat_e2, E);

    // ---------------- classification head ----------------
    head10<<<(M * 10 + 255) / 256, 256, 0, stream>>>(feat_e2, Wc, bc,
                                                     (float*)d_out, M);
    (void)out_size;
}
